// Block_5669356833485
// MI455X (gfx1250) — matrix operand order repaired and checked by reading, not compiled, not run on hardware
//
#include <hip/hip_runtime.h>
#include <cstdint>

// ---------------------------------------------------------------------------
// Types for CDNA5 WMMA (gfx1250, wave32)
// ---------------------------------------------------------------------------
typedef __bf16  v16bf __attribute__((ext_vector_type(16)));
typedef __bf16  v8bf  __attribute__((ext_vector_type(8)));
typedef float   v8f   __attribute__((ext_vector_type(8)));

union ABFrag { v16bf v; v8bf h[2]; };

// Async global->LDS 16B per lane (ASYNCcnt).  GV addressing: 64-bit vaddr, no saddr.
__device__ __forceinline__ void async_ld16(const void* gaddr, unsigned lds_off) {
  asm volatile("global_load_async_to_lds_b128 %0, %1, off"
               :: "v"(lds_off), "v"(gaddr)
               : "memory");
}
__device__ __forceinline__ void wait_async_le6() {
  asm volatile("s_wait_asynccnt 6" ::: "memory");
}
__device__ __forceinline__ void wait_async_0() {
  asm volatile("s_wait_asynccnt 0" ::: "memory");
}

// ---------------------------------------------------------------------------
// Elementwise helpers
// ---------------------------------------------------------------------------
__global__ void cvt_f32_to_bf16(const float* __restrict__ in, __bf16* __restrict__ out,
                                size_t n) {
  size_t i = (size_t)blockIdx.x * blockDim.x + threadIdx.x;
  if (i < n) out[i] = (__bf16)in[i];
}

// LayerNorm over last dim (C), output bf16.  One block per row.
__global__ __launch_bounds__(256)
void ln_kernel(const float* __restrict__ x, const float* __restrict__ g,
               const float* __restrict__ b, __bf16* __restrict__ out, int C) {
  const int row = blockIdx.x;
  const int tid = threadIdx.x;
  const float* xr = x + (size_t)row * C;
  float s = 0.f, s2 = 0.f;
  for (int c = tid; c < C; c += blockDim.x) {
    float v = xr[c];
    s += v; s2 += v * v;
  }
  __shared__ float sh[256], sh2[256];
  sh[tid] = s; sh2[tid] = s2;
  __syncthreads();
  for (int o = 128; o > 0; o >>= 1) {
    if (tid < o) { sh[tid] += sh[tid + o]; sh2[tid] += sh2[tid + o]; }
    __syncthreads();
  }
  const float mean = sh[0] / C;
  const float var  = sh2[0] / C - mean * mean;
  const float rstd = rsqrtf(var + 1e-5f);
  __bf16* orow = out + (size_t)row * C;
  for (int c = tid; c < C; c += blockDim.x) {
    float v = (xr[c] - mean) * rstd * g[c] + b[c];
    orow[c] = (__bf16)v;
  }
}

// time-shift mix:  y = xn[t]*mix + xn[t-1]*(1-mix)   (xn[-1] == 0)
__global__ void mix_kernel(const __bf16* __restrict__ xn,
                           const float* __restrict__ tmk,
                           const float* __restrict__ tmv,
                           const float* __restrict__ tmr,
                           __bf16* __restrict__ xk,
                           __bf16* __restrict__ xv,   // may be null
                           __bf16* __restrict__ xr,
                           int T, int C, size_t total) {
  size_t i = (size_t)blockIdx.x * blockDim.x + threadIdx.x;
  if (i >= total) return;
  int c = (int)(i % (size_t)C);
  size_t row = i / (size_t)C;
  int t = (int)(row % (size_t)T);
  float cur  = (float)xn[i];
  float prev = (t == 0) ? 0.f : (float)xn[i - C];
  float mk = tmk[c];
  xk[i] = (__bf16)(cur * mk + prev * (1.f - mk));
  if (xv != nullptr) {
    float mv = tmv[c];
    xv[i] = (__bf16)(cur * mv + prev * (1.f - mv));
  }
  float mr = tmr[c];
  xr[i] = (__bf16)(cur * mr + prev * (1.f - mr));
}

// RWKV-4 WKV linear recurrence, one thread per (b, c); fuses sr * y -> bf16.
__global__ __launch_bounds__(256)
void wkv_kernel(const float* __restrict__ time_decay,
                const float* __restrict__ time_first,
                const float* __restrict__ k, const float* __restrict__ v,
                const float* __restrict__ sr, __bf16* __restrict__ out,
                int T, int C, int B) {
  int idx = blockIdx.x * blockDim.x + threadIdx.x;
  if (idx >= B * C) return;
  int b = idx / C, c = idx % C;
  const float w = -__expf(time_decay[c]);
  const float u = time_first[c];
  float aa = 0.f, bb = 0.f, pp = -1e38f;
  size_t base = (size_t)b * T * C + c;
  for (int t = 0; t < T; ++t) {
    size_t i = base + (size_t)t * C;
    float kt = k[i], vt = v[i];
    float ww = u + kt;
    float q  = fmaxf(pp, ww);
    float e1 = __expf(pp - q);
    float e2 = __expf(ww - q);
    float y  = (e1 * aa + e2 * vt) / (e1 * bb + e2);
    out[i] = (__bf16)(sr[i] * y);
    float ww2 = pp + w;
    float q2  = fmaxf(ww2, kt);
    float e1b = __expf(ww2 - q2);
    float e2b = __expf(kt - q2);
    aa = e1b * aa + e2b * vt;
    bb = e1b * bb + e2b;
    pp = q2;
  }
}

// ---------------------------------------------------------------------------
// bf16 WMMA GEMM:  C[M,N] = A[M,K] @ W[N,K]^T   (A,W bf16 row-major, f32 acc)
// Block tile 128x256, 8 waves (2x4), wave tile 64x64 (4x4 wmma 16x16x32),
// K-step 32, double-buffered LDS filled via global_load_async_to_lds_b128.
// Steady state per wave per K-step: 6 async issues, 16 ds_load_b128, 16 wmma.
// Epilogues: 0 = f32 store, 1 = sigmoid f32, 2 = relu^2 -> bf16,
//            3 = res + acc -> f32, 4 = res + mul*acc -> f32.
// ---------------------------------------------------------------------------
#define BK   32
#define LDSW 40   // 32 + 8 pad halves; 80B row stride (16B aligned)

template <int EPI>
__global__ __launch_bounds__(256)
void gemm_bf16_wmma(const __bf16* __restrict__ A, const __bf16* __restrict__ W,
                    void* __restrict__ Out,
                    const float* __restrict__ res, const float* __restrict__ mul,
                    int M, int N, int K) {
  __shared__ __align__(16) __bf16 As[2][128][LDSW];   // 20 KB
  __shared__ __align__(16) __bf16 Bs[2][256][LDSW];   // 40 KB

  const int tid  = threadIdx.x;
  const int lane = tid & 31;
  const int wid  = tid >> 5;
  const int wm   = wid >> 2;       // 0..1  (64-row slabs)
  const int wn   = wid & 3;        // 0..3  (64-col slabs)
  const int l15  = lane & 15;
  const int lhi  = lane >> 4;      // 0/1 half-wave

  const int m0 = blockIdx.y * 128;
  const int n0 = blockIdx.x * 256;

  v8f acc[4][4];
#pragma unroll
  for (int mi = 0; mi < 4; ++mi)
#pragma unroll
    for (int nj = 0; nj < 4; ++nj)
      acc[mi][nj] = {};

  // --- async stage loader: A 512 chunks (2/thread), B 1024 chunks (4/thread) ---
  auto load_stage = [&](int buf, int k0) {
#pragma unroll
    for (int q = 0; q < 2; ++q) {
      int cid = tid * 2 + q;            // 0..511
      int row = cid >> 2;               // 0..127
      int kc  = cid & 3;                // 0..3 (8-half chunks)
      const __bf16* ga = A + (size_t)(m0 + row) * K + (k0 + kc * 8);
      unsigned la = (unsigned)(uintptr_t)&As[buf][row][kc * 8];
      async_ld16(ga, la);
    }
#pragma unroll
    for (int q = 0; q < 4; ++q) {
      int row = tid;                    // 0..255
      int kc  = q;                      // 0..3
      const __bf16* gw = W + (size_t)(n0 + row) * K + (k0 + kc * 8);
      unsigned lw = (unsigned)(uintptr_t)&Bs[buf][row][kc * 8];
      async_ld16(gw, lw);
    }
  };

  auto compute = [&](int buf) {
    // A fragments: lanes 0-15 row M=l15, halves 0-7 = K 0..7, 8-15 = K 16..23;
    //              lanes 16-31 same row, K offsets +8 / +24.
    ABFrag afr[4];
#pragma unroll
    for (int mi = 0; mi < 4; ++mi) {
      const __bf16* pa = &As[buf][wm * 64 + mi * 16 + l15][lhi * 8];
      afr[mi].h[0] = *(const v8bf*)pa;
      afr[mi].h[1] = *(const v8bf*)(pa + 16);
    }
    // B fragments: lane column N=l15; lanes 0-15 K 0..7,16..23, lanes 16-31 K 8..15,24..31.
    ABFrag bfr[4];
#pragma unroll
    for (int nj = 0; nj < 4; ++nj) {
      const __bf16* pb = &Bs[buf][wn * 64 + nj * 16 + l15][lhi * 8];
      bfr[nj].h[0] = *(const v8bf*)pb;
      bfr[nj].h[1] = *(const v8bf*)(pb + 16);
    }
#pragma unroll
    for (int mi = 0; mi < 4; ++mi)
#pragma unroll
      for (int nj = 0; nj < 4; ++nj)
        acc[mi][nj] = __builtin_amdgcn_wmma_f32_16x16x32_bf16(
            false, afr[mi].v, false, bfr[nj].v, (short)0, acc[mi][nj],
            false, false);
  };

  const int ksteps = K / BK;            // >= 32, so the peeled tail is safe
  load_stage(0, 0);
  for (int s = 0; s < ksteps - 1; ++s) {
    const int buf = s & 1;
    load_stage(buf ^ 1, (s + 1) * BK);
    wait_async_le6();                   // stage s (first 6/wave) complete
    __syncthreads();
    compute(buf);
    __syncthreads();
  }
  wait_async_0();
  __syncthreads();
  compute((ksteps - 1) & 1);

  // Epilogue.  C/D layout: VGPR r -> M = r (+8 for upper half-wave), N = lane&15.
#pragma unroll
  for (int mi = 0; mi < 4; ++mi) {
#pragma unroll
    for (int nj = 0; nj < 4; ++nj) {
      const int gn = n0 + wn * 64 + nj * 16 + l15;
#pragma unroll
      for (int r = 0; r < 8; ++r) {
        const int gm = m0 + wm * 64 + mi * 16 + r + lhi * 8;
        const size_t idx = (size_t)gm * N + gn;
        const float a = acc[mi][nj][r];
        if (EPI == 0) {
          ((float*)Out)[idx] = a;
        } else if (EPI == 1) {
          ((float*)Out)[idx] = 1.f / (1.f + __expf(-a));
        } else if (EPI == 2) {
          float t = a > 0.f ? a : 0.f;
          ((__bf16*)Out)[idx] = (__bf16)(t * t);
        } else if (EPI == 3) {
          ((float*)Out)[idx] = res[idx] + a;
        } else {
          ((float*)Out)[idx] = res[idx] + mul[idx] * a;
        }
      }
    }
  }
}

// ---------------------------------------------------------------------------
// Host-side orchestration
// ---------------------------------------------------------------------------
#define MiB(x) ((size_t)(x) << 20)

extern "C" void kernel_launch(void* const* d_in, const int* in_sizes, int n_in,
                              void* d_out, int out_size, void* d_ws, size_t ws_size,
                              hipStream_t stream) {
  (void)in_sizes; (void)n_in; (void)out_size; (void)ws_size;
  const int Bn = 8, T = 2048, C = 1024, HID = 4096;
  const int MT = Bn * T;               // 16384 rows

  const float* x     = (const float*)d_in[0];
  const float* ln1_g = (const float*)d_in[1];
  const float* ln1_b = (const float*)d_in[2];
  const float* ln2_g = (const float*)d_in[3];
  const float* ln2_b = (const float*)d_in[4];
  const float* td    = (const float*)d_in[5];
  const float* tf    = (const float*)d_in[6];
  const float* tmk   = (const float*)d_in[7];
  const float* tmv   = (const float*)d_in[8];
  const float* tmr   = (const float*)d_in[9];
  const float* Wk    = (const float*)d_in[10];
  const float* Wv    = (const float*)d_in[11];
  const float* Wr    = (const float*)d_in[12];
  const float* Wo    = (const float*)d_in[13];
  const float* f_tmk = (const float*)d_in[14];
  const float* f_tmr = (const float*)d_in[15];
  const float* fWk   = (const float*)d_in[16];
  const float* fWr   = (const float*)d_in[17];
  const float* fWv   = (const float*)d_in[18];
  float* out = (float*)d_out;

  // --- workspace layout (phase-overlapped) ---
  char* ws = (char*)d_ws;
  __bf16* xn   = (__bf16*)(ws + MiB(0));     // 32 MiB  (MT*C bf16)
  __bf16* xk   = (__bf16*)(ws + MiB(32));
  __bf16* xv   = (__bf16*)(ws + MiB(64));
  __bf16* xr   = (__bf16*)(ws + MiB(96));
  float*  kf   = (float* )(ws + MiB(128));   // 64 MiB
  float*  vf   = (float* )(ws + MiB(192));
  float*  srf  = (float* )(ws + MiB(256));   // reused as s2 in FFN
  __bf16* abf  = (__bf16*)(ws + MiB(320));   // sr*y  (32 MiB)
  __bf16* kkbf = (__bf16*)(ws + MiB(128));   // FFN kk, 128 MiB, reuses k/v
  __bf16* wk_b = (__bf16*)(ws + MiB(352));
  __bf16* wv_b = (__bf16*)(ws + MiB(354));
  __bf16* wr_b = (__bf16*)(ws + MiB(356));
  __bf16* wo_b = (__bf16*)(ws + MiB(358));
  __bf16* fwk_b= (__bf16*)(ws + MiB(360));   // 8 MiB
  __bf16* fwr_b= (__bf16*)(ws + MiB(368));
  __bf16* fwv_b= (__bf16*)(ws + MiB(370));   // 8 MiB -> ends 378 MiB

  auto cvt = [&](const float* s, __bf16* d, size_t n) {
    cvt_f32_to_bf16<<<dim3((unsigned)((n + 255) / 256)), dim3(256), 0, stream>>>(s, d, n);
  };
  cvt(Wk,  wk_b,  (size_t)C * C);
  cvt(Wv,  wv_b,  (size_t)C * C);
  cvt(Wr,  wr_b,  (size_t)C * C);
  cvt(Wo,  wo_b,  (size_t)C * C);
  cvt(fWk, fwk_b, (size_t)HID * C);
  cvt(fWr, fwr_b, (size_t)C * C);
  cvt(fWv, fwv_b, (size_t)C * HID);

  const size_t total = (size_t)MT * C;
  const dim3 blk(256);
  const dim3 gEl((unsigned)((total + 255) / 256));
  const dim3 gCC(C / 256, MT / 128);        // (4, 128)
  const dim3 gFK(HID / 256, MT / 128);      // (16, 128)

  // ---- time-mix branch ----
  ln_kernel<<<dim3(MT), blk, 0, stream>>>(x, ln1_g, ln1_b, xn, C);
  mix_kernel<<<gEl, blk, 0, stream>>>(xn, tmk, tmv, tmr, xk, xv, xr, T, C, total);
  gemm_bf16_wmma<0><<<gCC, blk, 0, stream>>>(xk, wk_b, kf,  nullptr, nullptr, MT, C, C);
  gemm_bf16_wmma<0><<<gCC, blk, 0, stream>>>(xv, wv_b, vf,  nullptr, nullptr, MT, C, C);
  gemm_bf16_wmma<1><<<gCC, blk, 0, stream>>>(xr, wr_b, srf, nullptr, nullptr, MT, C, C);
  wkv_kernel<<<dim3((Bn * C) / 256), blk, 0, stream>>>(td, tf, kf, vf, srf, abf, T, C, Bn);
  gemm_bf16_wmma<3><<<gCC, blk, 0, stream>>>(abf, wo_b, out, x, nullptr, MT, C, C);

  // ---- channel-mix (FFN) branch ----
  ln_kernel<<<dim3(MT), blk, 0, stream>>>(out, ln2_g, ln2_b, xn, C);
  mix_kernel<<<gEl, blk, 0, stream>>>(xn, f_tmk, nullptr, f_tmr, xk, nullptr, xr, T, C, total);
  gemm_bf16_wmma<2><<<gFK, blk, 0, stream>>>(xk, fwk_b, kkbf, nullptr, nullptr, MT, HID, C);
  gemm_bf16_wmma<1><<<gCC, blk, 0, stream>>>(xr, fwr_b, srf,  nullptr, nullptr, MT, C, C);
  gemm_bf16_wmma<4><<<gCC, blk, 0, stream>>>(kkbf, fwv_b, out, out, srf, MT, C, HID);
}
